// EnhancedRNN_64613488001026
// MI455X (gfx1250) — matrix operand order repaired and checked by reading, not compiled, not run on hardware
//
#include <hip/hip_runtime.h>
#include <hip/hip_bf16.h>
#include <math.h>

typedef __bf16 bf16;
typedef __attribute__((ext_vector_type(16))) __bf16 v16bf;
typedef __attribute__((ext_vector_type(8)))  __bf16 v8bf;
typedef __attribute__((ext_vector_type(8)))  float  v8f;

#define D_MODEL 1024
#define GDIM    3072      // 3 * D_MODEL
#define BATCH   32
#define SEQT    512
#define ROWS    (BATCH * SEQT)   // 16384
#define VOCABSZ 256
#define NWG_SCAN 16

// ---------------------------------------------------------------------------
// WMMA fragment loaders (bf16, 16x16x32). Row-major source matrices.
// A (16x32, MxK): lane holds row m = m0+(lane&15); 16 consecutive K values
//   starting at k0 + ((lane>=16)?16:0).  -> one 32-byte load.
// B (32x16, KxN): lane holds column n = n0+(lane&15); 16 consecutive K values
//   starting at k0 + ((lane>=16)?16:0).  -> one 32-byte load from W[n][k].
// C/D (16x16 f32): elem r <-> (m = m0 + r + ((lane>=16)?8:0), n = n0+(lane&15))
// ---------------------------------------------------------------------------
__device__ __forceinline__ v16bf load_a_frag(const bf16* __restrict__ base,
                                             int lane, int ldk, int m, int k0) {
    const bf16* p = base + (size_t)m * ldk + (size_t)k0 + ((lane >= 16) ? 16 : 0);
    return *(const v16bf*)p;
}

__device__ __forceinline__ v16bf load_b_frag(const bf16* __restrict__ wbase,
                                             int lane, int ldk, int n, int k0) {
    const bf16* p = wbase + (size_t)n * ldk + (size_t)k0 + ((lane >= 16) ? 16 : 0);
    return *(const v16bf*)p;
}

__device__ __forceinline__ v8f wmma_bf16(v16bf a, v16bf b, v8f c) {
    return __builtin_amdgcn_wmma_f32_16x16x32_bf16(false, a, false, b,
                                                   (short)0, c, false, false);
}

// ---------------------------------------------------------------------------
// Generic bf16 WMMA GEMM:  out[M,N] = A[M,K] * Bw[N,K]^T (+ bias[n])
// WG = 256 threads = 8 waves arranged 2(M) x 4(N); each wave: 64x64 tile
// (4 M-tiles x 4 N-tiles of 16x16).  Grid: (N/256, M/128).
// ---------------------------------------------------------------------------
template<bool OUT_BF16, bool ADD_BIAS>
__global__ __launch_bounds__(256) void gemm_bf16_wmma(
    const bf16* __restrict__ A, const bf16* __restrict__ Bw,
    const float* __restrict__ bias, void* __restrict__ outp,
    int M, int N, int K)
{
    const int lane = threadIdx.x & 31;
    const int w    = threadIdx.x >> 5;
    const int wm   = w >> 2;        // 0..1
    const int wn   = w & 3;         // 0..3
    const int m_base = blockIdx.y * 128 + wm * 64;
    const int n_base = blockIdx.x * 256 + wn * 64;
    const int l15 = lane & 15;

    v8f acc[4][4];
    #pragma unroll
    for (int mt = 0; mt < 4; ++mt)
        #pragma unroll
        for (int nt = 0; nt < 4; ++nt)
            acc[mt][nt] = (v8f){0.f,0.f,0.f,0.f,0.f,0.f,0.f,0.f};

    for (int k0 = 0; k0 < K; k0 += 32) {
        v16bf a[4];
        #pragma unroll
        for (int mt = 0; mt < 4; ++mt)
            a[mt] = load_a_frag(A, lane, K, m_base + mt * 16 + l15, k0);
        #pragma unroll
        for (int nt = 0; nt < 4; ++nt) {
            v16bf bfrag = load_b_frag(Bw, lane, K, n_base + nt * 16 + l15, k0);
            #pragma unroll
            for (int mt = 0; mt < 4; ++mt)
                acc[mt][nt] = wmma_bf16(a[mt], bfrag, acc[mt][nt]);
        }
    }

    const int mrow_off = (lane >= 16) ? 8 : 0;
    #pragma unroll
    for (int nt = 0; nt < 4; ++nt) {
        int n = n_base + nt * 16 + l15;
        float bv = ADD_BIAS ? bias[n] : 0.0f;
        #pragma unroll
        for (int mt = 0; mt < 4; ++mt) {
            #pragma unroll
            for (int r = 0; r < 8; ++r) {
                int m = m_base + mt * 16 + r + mrow_off;
                float v = acc[mt][nt][r] + bv;
                if (OUT_BF16) ((bf16*)outp)[(size_t)m * N + n] = (bf16)v;
                else          ((float*)outp)[(size_t)m * N + n] = v;
            }
        }
    }
}

// ---------------------------------------------------------------------------
// Persistent GRU recurrent scan for one layer.
// Grid = 16 WGs x 256 threads (8 waves), all co-resident; device-wide spin
// barrier per timestep (monotonic counter).
// Per WG: 4 feature-tiles of 16 features. Per wave: one feature-tile,
// both M-tiles (batch 0..15 and 16..31), all three gates (r,z,n) -> 6
// accumulators share A-fragments; K=1024 split in halves across wave pairs,
// partials combined through LDS.  hg = h * Whh^T; gates in registers.
// ---------------------------------------------------------------------------
__global__ __launch_bounds__(256) void gru_scan_kernel(
    const bf16* __restrict__ xg,     // [ROWS, 3D], bih already baked in
    const bf16* __restrict__ Whh,    // [3D, D] row-major
    const float* __restrict__ bhh,   // [3D]
    bf16* __restrict__ seq_out,      // [ROWS, D]
    bf16* __restrict__ hbuf,         // [2, BATCH, D], zero-initialized
    unsigned* __restrict__ counter)  // zero-initialized
{
    __shared__ float lds[4][6][256];

    const int tid   = threadIdx.x;
    const int lane  = tid & 31;
    const int w     = tid >> 5;
    const int ftl   = w & 3;                  // feature tile within WG
    const int khalf = w >> 2;                 // K-split half: 0 or 1
    const int ft    = blockIdx.x * 4 + ftl;   // 0..63 global feature tile
    const int cbase = ft * 16;
    const int l15   = lane & 15;
    const int c     = cbase + l15;            // feature column 0..1023
    const int mrow_off = (lane >= 16) ? 8 : 0;

    const float bhr = bhh[c];
    const float bhz = bhh[1024 + c];
    const float bhn = bhh[2048 + c];

    const int kb = khalf * 512;
    const int ke = kb + 512;

    for (int t = 0; t < SEQT; ++t) {
        const bf16* hcur = hbuf + (size_t)(t & 1) * BATCH * D_MODEL;
        bf16*       hnxt = hbuf + (size_t)((t + 1) & 1) * BATCH * D_MODEL;

        v8f acc[6];   // [mt*3 + gate], mt in {0,1}, gate in {r,z,n}
        #pragma unroll
        for (int i = 0; i < 6; ++i)
            acc[i] = (v8f){0.f,0.f,0.f,0.f,0.f,0.f,0.f,0.f};

        for (int k0 = kb; k0 < ke; k0 += 32) {
            v16bf a0 = load_a_frag(hcur, lane, D_MODEL, l15,      k0);
            v16bf a1 = load_a_frag(hcur, lane, D_MODEL, 16 + l15, k0);
            #pragma unroll
            for (int g = 0; g < 3; ++g) {
                v16bf bfrag = load_b_frag(Whh, lane, D_MODEL,
                                          g * 1024 + cbase + l15, k0);
                acc[g]     = wmma_bf16(a0, bfrag, acc[g]);
                acc[3 + g] = wmma_bf16(a1, bfrag, acc[3 + g]);
            }
        }

        if (khalf == 1) {
            #pragma unroll
            for (int i = 0; i < 6; ++i)
                #pragma unroll
                for (int r = 0; r < 8; ++r)
                    lds[ftl][i][r * 32 + lane] = acc[i][r];
        }
        __syncthreads();

        if (khalf == 0) {
            #pragma unroll
            for (int i = 0; i < 6; ++i)
                #pragma unroll
                for (int r = 0; r < 8; ++r)
                    acc[i][r] += lds[ftl][i][r * 32 + lane];

            #pragma unroll
            for (int mt = 0; mt < 2; ++mt) {
                #pragma unroll
                for (int r = 0; r < 8; ++r) {
                    int bi = mt * 16 + r + mrow_off;               // batch row
                    const bf16* xrow = xg + ((size_t)bi * SEQT + t) * GDIM;
                    float xr = (float)xrow[c];
                    float xz = (float)xrow[1024 + c];
                    float xn = (float)xrow[2048 + c];
                    float hr = acc[mt * 3 + 0][r] + bhr;
                    float hz = acc[mt * 3 + 1][r] + bhz;
                    float hn = acc[mt * 3 + 2][r] + bhn;
                    float rg = 1.f / (1.f + __expf(-(xr + hr)));
                    float zg = 1.f / (1.f + __expf(-(xz + hz)));
                    float ng = tanhf(xn + rg * hn);
                    float hold = (float)hcur[(size_t)bi * D_MODEL + c];
                    float hnew = (1.f - zg) * ng + zg * hold;
                    bf16 hb = (bf16)hnew;
                    hnxt[(size_t)bi * D_MODEL + c] = hb;
                    seq_out[((size_t)bi * SEQT + t) * D_MODEL + c] = hb;
                }
            }
        }

        // release h writes, then device-wide barrier (monotonic counter)
        __builtin_amdgcn_fence(__ATOMIC_RELEASE, "agent");
        __syncthreads();
        if (tid == 0) {
            __hip_atomic_fetch_add(counter, 1u, __ATOMIC_ACQ_REL,
                                   __HIP_MEMORY_SCOPE_AGENT);
            unsigned target = (unsigned)(t + 1) * NWG_SCAN;
            while (__hip_atomic_load(counter, __ATOMIC_ACQUIRE,
                                     __HIP_MEMORY_SCOPE_AGENT) < target)
                __builtin_amdgcn_s_sleep(2);
        }
        __syncthreads();
        __builtin_amdgcn_fence(__ATOMIC_ACQUIRE, "agent");
    }
}

// ---------------------------------------------------------------------------
// LayerNorm over D=1024, in place on bf16 rows. One WG (256 thr) per row.
// ---------------------------------------------------------------------------
__global__ __launch_bounds__(256) void layernorm_kernel(
    bf16* __restrict__ h, const float* __restrict__ gamma,
    const float* __restrict__ beta)
{
    __shared__ float s1[8], s2[8];
    const int tid = threadIdx.x;
    bf16* row = h + (size_t)blockIdx.x * D_MODEL;
    float v[4]; float a = 0.f, b = 0.f;
    #pragma unroll
    for (int i = 0; i < 4; ++i) {
        v[i] = (float)row[tid + i * 256];
        a += v[i]; b += v[i] * v[i];
    }
    #pragma unroll
    for (int off = 16; off > 0; off >>= 1) {
        a += __shfl_down(a, off, 32);
        b += __shfl_down(b, off, 32);
    }
    if ((tid & 31) == 0) { s1[tid >> 5] = a; s2[tid >> 5] = b; }
    __syncthreads();
    if (tid == 0) {
        float ta = 0.f, tb = 0.f;
        for (int i = 0; i < 8; ++i) { ta += s1[i]; tb += s2[i]; }
        s1[0] = ta; s2[0] = tb;
    }
    __syncthreads();
    float mu   = s1[0] * (1.f / D_MODEL);
    float var  = s2[0] * (1.f / D_MODEL) - mu * mu;
    float rstd = rsqrtf(var + 1e-5f);
    #pragma unroll
    for (int i = 0; i < 4; ++i) {
        int cI = tid + i * 256;
        float o = (v[i] - mu) * rstd * gamma[cI] + beta[cI];
        row[cI] = (bf16)o;
    }
}

// ---------------------------------------------------------------------------
// Embedding gather (bf16 table), 8 elements per thread.
// ---------------------------------------------------------------------------
__global__ __launch_bounds__(256) void embed_kernel(
    const int* __restrict__ x, const bf16* __restrict__ Ebf,
    bf16* __restrict__ out)
{
    size_t i = (size_t)blockIdx.x * 256 + threadIdx.x;   // v8bf chunk id
    size_t rowI = i >> 7;                                 // 128 chunks / row
    int chunk   = (int)(i & 127);
    int tok = x[rowI];
    const v8bf* src = (const v8bf*)(Ebf + (size_t)tok * D_MODEL);
    v8bf*       dst = (v8bf*)(out + rowI * D_MODEL);
    dst[chunk] = src[chunk];
}

__global__ __launch_bounds__(256) void f32_to_bf16_kernel(
    const float* __restrict__ in, bf16* __restrict__ out, size_t n)
{
    size_t i = (size_t)blockIdx.x * blockDim.x + threadIdx.x;
    size_t stride = (size_t)gridDim.x * blockDim.x;
    for (; i < n; i += stride) out[i] = (bf16)in[i];
}

// ---------------------------------------------------------------------------
extern "C" void kernel_launch(void* const* d_in, const int* in_sizes, int n_in,
                              void* d_out, int out_size, void* d_ws, size_t ws_size,
                              hipStream_t stream)
{
    const int*   x     = (const int*)d_in[0];    // [32,512]
    const float* E     = (const float*)d_in[1];  // [256,1024]
    const float* Wih   = (const float*)d_in[2];  // [2,3072,1024]
    const float* Whh   = (const float*)d_in[3];  // [2,3072,1024]
    const float* bih   = (const float*)d_in[4];  // [2,3072]
    const float* bhh   = (const float*)d_in[5];  // [2,3072]
    const float* gamma = (const float*)d_in[6];  // [1024]
    const float* beta  = (const float*)d_in[7];  // [1024]
    float* logits = (float*)d_out;               // [32,512,256] f32

    char* ws = (char*)d_ws;
    size_t off = 0;
    auto take = [&](size_t bytes) -> char* {
        char* p = ws + off;
        off = (off + bytes + 255) & ~(size_t)255;
        return p;
    };
    bf16*     Ebf   = (bf16*)take((size_t)VOCABSZ * D_MODEL * 2);   // 0.5 MB
    bf16*     WihBf = (bf16*)take((size_t)2 * GDIM * D_MODEL * 2);  // 12 MB
    bf16*     WhhBf = (bf16*)take((size_t)2 * GDIM * D_MODEL * 2);  // 12 MB
    bf16*     Xg    = (bf16*)take((size_t)ROWS * GDIM * 2);         // 96 MB
    bf16*     SeqA  = (bf16*)take((size_t)ROWS * D_MODEL * 2);      // 32 MB
    bf16*     SeqB  = (bf16*)take((size_t)ROWS * D_MODEL * 2);      // 32 MB
    bf16*     Hbuf  = (bf16*)take((size_t)2 * BATCH * D_MODEL * 2); // 128 KB
    unsigned* Cnt   = (unsigned*)take(256);                         // barriers

    const size_t hbufBytes = (size_t)2 * BATCH * D_MODEL * 2;
    // zero h0 and both barrier counters (in-graph, so replays reset them)
    hipMemsetAsync(Hbuf, 0, hbufBytes + 256, stream);

    f32_to_bf16_kernel<<<2048, 256, 0, stream>>>(E, Ebf, (size_t)VOCABSZ * D_MODEL);
    f32_to_bf16_kernel<<<4096, 256, 0, stream>>>(Wih, WihBf, (size_t)2 * GDIM * D_MODEL);
    f32_to_bf16_kernel<<<4096, 256, 0, stream>>>(Whh, WhhBf, (size_t)2 * GDIM * D_MODEL);
    embed_kernel<<<(ROWS * D_MODEL / 8) / 256, 256, 0, stream>>>(x, Ebf, SeqA);

    // ---- layer 0 ----
    gemm_bf16_wmma<true, true><<<dim3(GDIM / 256, ROWS / 128), 256, 0, stream>>>(
        SeqA, WihBf, bih, Xg, ROWS, GDIM, D_MODEL);
    gru_scan_kernel<<<NWG_SCAN, 256, 0, stream>>>(
        Xg, WhhBf, bhh, SeqB, Hbuf, Cnt);

    // ---- layer 1 ----
    hipMemsetAsync(Hbuf, 0, hbufBytes, stream);   // h0 = 0 again
    gemm_bf16_wmma<true, true><<<dim3(GDIM / 256, ROWS / 128), 256, 0, stream>>>(
        SeqB, WihBf + (size_t)GDIM * D_MODEL, bih + GDIM, Xg, ROWS, GDIM, D_MODEL);
    gru_scan_kernel<<<NWG_SCAN, 256, 0, stream>>>(
        Xg, WhhBf + (size_t)GDIM * D_MODEL, bhh + GDIM, SeqA, Hbuf, Cnt + 32);

    // ---- LayerNorm + tied head ----
    layernorm_kernel<<<ROWS, 256, 0, stream>>>(SeqA, gamma, beta);
    gemm_bf16_wmma<false, false><<<dim3(VOCABSZ / 256, ROWS / 128), 256, 0, stream>>>(
        SeqA, Ebf, nullptr, logits, ROWS, VOCABSZ, D_MODEL);

    (void)in_sizes; (void)n_in; (void)out_size; (void)ws_size;
}
